// VQGAN_4037269259122
// MI455X (gfx1250) — compile-verified, matrix operand order checked by reading, not run on hardware
//
#include <hip/hip_runtime.h>
#include <hip/hip_bf16.h>
#include <stdint.h>

// ---------------------------------------------------------------------------
// CDNA5 (gfx1250) VQ-GAN forward.
//  * Convs (fwd + transposed-as-4-parity-subconvs) = implicit GEMMs on
//    v_wmma_f32_16x16x32_f16, f32 accumulate, 64-wide K steps.
//  * Weight tiles pre-packed fragment-ready; moved by the Tensor Data Mover
//    as a 2-D tile (2 k-slabs per issue), synced with s_wait_tensorcnt.
//  * Branchless batched im2col gather (16 loads in flight per thread).
//  * All spatial decodes are shift/mask (power-of-two dims) - no int division.
// ---------------------------------------------------------------------------

typedef __attribute__((ext_vector_type(16))) _Float16 v16h;
typedef __attribute__((ext_vector_type(8)))  float    v8f;
typedef unsigned int u32x4 __attribute__((ext_vector_type(4)));
typedef int          i32x4 __attribute__((ext_vector_type(4)));
typedef int          i32x8 __attribute__((ext_vector_type(8)));

#define DEV __device__ __forceinline__

// -------------------- WMMA helpers ------------------------------------------
DEV v8f wmma_f16(v16h a, v16h b, v8f c) {
  return __builtin_amdgcn_wmma_f32_16x16x32_f16(
      false, a, false, b, (short)0, c, false, false);
}

// A tile in LDS: [m][72] f16 (144B row stride: 36*r mod 64 banks all distinct
// for r=0..15 -> conflict-free b128 fragment loads).  Slab s' at column s'*32.
// ISA 7.12.2 16-bit A 16x32: lanes 0-15 row=lane, K {0..7,16..23}; lanes
// 16-31 same rows, K {8..15,24..31}.
DEV v16h ldsFragA(const _Float16* __restrict__ base, int lane) {
  const int row = lane & 15, half = lane >> 4;
  const uint32_t* rp = (const uint32_t*)(base + row * 72 + half * 8);
  union { v16h v; uint32_t u[8]; } f;
#pragma unroll
  for (int p = 0; p < 8; ++p) f.u[p] = rp[(p < 4) ? p : (p + 4)];
  return f.v;
}

// B chunk in LDS is fragment-ready: lane's 8 dwords are contiguous.
DEV v16h ldsFragFR(const _Float16* __restrict__ chunk, int lane) {
  const uint32_t* rp = (const uint32_t*)chunk + lane * 8;
  union { v16h v; uint32_t u[8]; } f;
#pragma unroll
  for (int p = 0; p < 8; ++p) f.u[p] = rp[p];
  return f.v;
}

DEV float act_apply(float v, int ACT) {
  if (ACT == 1) return v > 0.f ? v : 0.2f * v;
  if (ACT == 2) return tanhf(v);
  return v;
}

// -------------------- Tensor Data Mover: 2-D B tile -> LDS -------------------
// Fragment-ready pack: chunk (kslab, nb) = 1KB contiguous. One issue moves
// tile rows = 2 k-slabs x 4KB (tile_dim0=1024 dw, tile_dim1=2), row stride =
// NBtot KB.  tensor_dim0/1 clamp => OOB chunks/slabs arrive as zeros.
DEV void tdm_load_B(const _Float16* __restrict__ gbase, int slab, int nb0,
                    int NBtot, int nslabs, _Float16* lds) {
  uint64_t ga = (uint64_t)(uintptr_t)gbase +
                ((uint64_t)slab * NBtot + nb0) * 1024u;
  uint32_t ldsOff = (uint32_t)(uintptr_t)lds;     // low 32 bits = LDS offset
  uint32_t dim0 = (uint32_t)(NBtot - nb0) * 256u; // dwords valid in a row
  uint32_t dim1 = (uint32_t)(nslabs - slab);      // k-slab rows remaining
  uint32_t str0 = (uint32_t)NBtot * 256u;         // row stride (dwords)
  u32x4 g0 = { 1u,                                 // count=1 (user D#)
               ldsOff,                             // lds_addr
               (uint32_t)ga,                       // global_addr[31:0]
               (uint32_t)((ga >> 32) & 0x1FFFFFFull) | (2u << 30) }; // type=2
  i32x8 g1 = { (int)(2u << 16),                    // data_size = 4B
               (int)((dim0 & 0xFFFFu) << 16),      // tensor_dim0[15:0]
               (int)((dim0 >> 16) | ((dim1 & 0xFFFFu) << 16)),
               (int)((dim1 >> 16) | (1024u << 16)),// tile_dim0 = 1024 dw
               2,                                  // tile_dim1 = 2
               (int)str0, 0, 0 };                  // tensor_dim0_stride
  i32x4 gz = { 0, 0, 0, 0 };
#if defined(__clang_major__) && __clang_major__ >= 23
  i32x8 g4 = { 0, 0, 0, 0, 0, 0, 0, 0 };
  __builtin_amdgcn_tensor_load_to_lds(g0, g1, gz, gz, g4, 0);
#else
  __builtin_amdgcn_tensor_load_to_lds(g0, g1, gz, gz, 0);
#endif
}

// -------------------- fragment-ready weight packing --------------------------
DEV void fr_decode(int e, int NBtot, int& kg, int& n) {
  int lo = e & 1, p = (e >> 1) & 7, lane = (e >> 4) & 31, chunk = e >> 9;
  int nb = chunk % NBtot, kslab = chunk / NBtot;
  int c = lane & 15, grp = lane >> 4;
  int kr = ((p >> 2) << 4) + (grp << 3) + ((p & 3) << 1) + lo;
  kg = (kslab << 5) + kr;
  n = (nb << 4) + c;
}

__global__ void pack_conv_fr(const float* __restrict__ w, _Float16* __restrict__ pk,
                             int K, int Cout, int NBtot, int total) {
  int e = blockIdx.x * blockDim.x + threadIdx.x;
  if (e >= total) return;
  int kg, n; fr_decode(e, NBtot, kg, n);
  pk[e] = (kg < K && n < Cout) ? (_Float16)w[(size_t)n * K + kg] : (_Float16)0.f;
}

__global__ void pack_convt_fr(const float* __restrict__ w, _Float16* __restrict__ pk,
                              int Cin, int Cout, int NBtot, int perPar) {
  int e = blockIdx.x * blockDim.x + threadIdx.x;
  if (e >= 4 * perPar) return;
  int par = e / perPar, r = e - par * perPar;
  int kg, n; fr_decode(r, NBtot, kg, n);
  int K = Cin * 4;
  _Float16 v = (_Float16)0.f;
  if (kg < K && n < Cout) {
    int ci = kg >> 2, th = (kg >> 1) & 1, tw = kg & 1;
    int ph = par >> 1, pw = par & 1;
    int kh = ph ? (th ? 2 : 0) : (th ? 3 : 1);
    int kw = pw ? (tw ? 2 : 0) : (tw ? 3 : 1);
    v = (_Float16)w[((size_t)ci * Cout + n) * 16 + kh * 4 + kw];
  }
  pk[e] = v;
}

__global__ void cb_prep(const float* __restrict__ cb, _Float16* __restrict__ cbh,
                        float* __restrict__ cbn, float* __restrict__ lossAcc) {
  int idx = blockIdx.x * blockDim.x + threadIdx.x;
  if (idx < 512 * 256) cbh[idx] = (_Float16)cb[idx];
  if (idx < 512) {
    float s = 0.f;
    const float* r = cb + (size_t)idx * 256;
#pragma unroll 8
    for (int d = 0; d < 256; ++d) s += r[d] * r[d];
    cbn[idx] = s;
  }
  if (idx == 0) *lossAcc = 0.f;
}

// -------------------- implicit-GEMM conv (fwd & transposed) ------------------
// lgHW/lgWo: log2 of the decode image (full image MODE0, parity sub-image
// MODE1).  All spatial sizes are powers of two.
template <int MODE, int ACT, typename TIN, typename TOUT>
__global__ __launch_bounds__(256) void conv_gemm(
    const TIN* __restrict__ in, const _Float16* __restrict__ wpackBase,
    const float* __restrict__ bias, TOUT* __restrict__ out,
    int Cin, int Hi, int Wi, int Cout, int Ho, int Wo, int K, int NBtot,
    int lgHW, int lgWo) {
  __shared__ _Float16 sA[2][64 * 72];
  __shared__ _Float16 sB[2][4096];

  const int tid = threadIdx.x, lane = tid & 31, wave = tid >> 5;
  const int wm = wave & 3, wn = wave >> 2;
  const int m0 = blockIdx.x * 64, n0 = blockIdx.y * 64, nb0 = blockIdx.y * 4;
  const int nslabs = (K + 31) >> 5;
  const int niter = (nslabs + 1) >> 1;

  int ph = 0, pw = 0;
  const _Float16* wp = wpackBase;
  if (MODE == 1) {
    ph = blockIdx.z >> 1; pw = blockIdx.z & 1;
    wp += (size_t)blockIdx.z * nslabs * NBtot * 512;
  }
  const int HiWi = Hi * Wi;
  const int mskHW = (1 << lgHW) - 1, mskWo = (1 << lgWo) - 1;

  // This thread owns one A row (m) and 16 consecutive K elements per iter.
  const int mlRow = tid >> 2;
  const int klBase = (tid & 3) << 4;
  int preOff, preH, preW;
  {
    int m = m0 + mlRow;
    int b = m >> lgHW, rr = m & mskHW;
    int oh = rr >> lgWo, ow = rr & mskWo;
    preOff = b * Cin * HiWi;
    if (MODE == 0) { preH = 2 * oh - 1; preW = 2 * ow - 1; }
    else           { preH = oh;         preW = ow; }
  }

  // Branchless batched im2col gather: 16 unconditional loads, then select.
  auto gatherA = [&](int ks, int buf) {
    int addrs[16], okm[16];
    TIN raw[16];
#pragma unroll
    for (int i = 0; i < 16; ++i) {
      int kg = ks + klBase + i;
      int ci, ih, iw;
      if (MODE == 0) {
        ci = kg >> 4; ih = preH + ((kg >> 2) & 3); iw = preW + (kg & 3);
      } else {
        ci = kg >> 2; int th = (kg >> 1) & 1, tw = kg & 1;
        ih = preH + (ph ? 1 - th : -th);
        iw = preW + (pw ? 1 - tw : -tw);
      }
      int ok = (kg < K) & ((unsigned)ih < (unsigned)Hi) & ((unsigned)iw < (unsigned)Wi);
      addrs[i] = ok ? (preOff + ci * HiWi + ih * Wi + iw) : 0;
      okm[i] = ok;
    }
#pragma unroll
    for (int i = 0; i < 16; ++i) raw[i] = in[addrs[i]];
    union { _Float16 h[8]; uint4 q; } u0, u1;
#pragma unroll
    for (int i = 0; i < 8; ++i) {
      u0.h[i] = okm[i] ? (_Float16)(float)raw[i] : (_Float16)0.f;
      u1.h[i] = okm[i + 8] ? (_Float16)(float)raw[i + 8] : (_Float16)0.f;
    }
    _Float16* dst = &sA[buf][mlRow * 72 + klBase];
    *(uint4*)dst = u0.q;                 // two aligned b128 stores
    *(uint4*)(dst + 8) = u1.q;
  };

  v8f acc0 = {}, acc1 = {};

  gatherA(0, 0);
  if (wave == 0) {
    tdm_load_B(wp, 0, nb0, NBtot, nslabs, &sB[0][0]);
    __builtin_amdgcn_s_wait_tensorcnt(0);
  }
  __syncthreads();

  int cur = 0;
  for (int it = 0; it < niter; ++it) {
    int nxt = cur ^ 1;
    if (it + 1 < niter) {
      gatherA((it + 1) * 64, nxt);
      if (wave == 0)
        tdm_load_B(wp, (it + 1) * 2, nb0, NBtot, nslabs, &sB[nxt][0]);
    }
    const _Float16* Ab = &sA[cur][wm * 16 * 72];
    v16h a0  = ldsFragA(Ab, lane);
    v16h a1  = ldsFragA(Ab + 32, lane);
    v16h b00 = ldsFragFR(&sB[cur][(wn * 2 + 0) * 512], lane);
    v16h b01 = ldsFragFR(&sB[cur][(wn * 2 + 1) * 512], lane);
    v16h b10 = ldsFragFR(&sB[cur][(4 + wn * 2 + 0) * 512], lane);
    v16h b11 = ldsFragFR(&sB[cur][(4 + wn * 2 + 1) * 512], lane);
    acc0 = wmma_f16(a0, b00, acc0);
    acc1 = wmma_f16(a0, b01, acc1);
    acc0 = wmma_f16(a1, b10, acc0);
    acc1 = wmma_f16(a1, b11, acc1);
    if (wave == 0) __builtin_amdgcn_s_wait_tensorcnt(0);
    __syncthreads();
    cur = nxt;
  }

  // Store (C/D layout: VGPR r -> M = r (+8 for lane half), N = lane&15).
  const int half = lane >> 4, nl = lane & 15;
#pragma unroll
  for (int part = 0; part < 2; ++part) {
    v8f accv = part ? acc1 : acc0;
    int n = n0 + wn * 32 + part * 16 + nl;
    if (n >= Cout) continue;
    float bsv = bias[n];
#pragma unroll
    for (int r = 0; r < 8; ++r) {
      int m = m0 + wm * 16 + r + 8 * half;
      float v = act_apply(accv[r] + bsv, ACT);
      int b = m >> lgHW, rr = m & mskHW;
      int oh, ow;
      if (MODE == 0) { oh = rr >> lgWo; ow = rr & mskWo; }
      else { oh = 2 * (rr >> lgWo) + ph; ow = 2 * (rr & mskWo) + pw; }
      out[((size_t)(b * Cout + n) * Ho + oh) * Wo + ow] = (TOUT)v;
    }
  }
}

// -------------------- vector quantization ------------------------------------
__global__ __launch_bounds__(256) void vq_kernel(
    const _Float16* __restrict__ z, const _Float16* __restrict__ cbh,
    const float* __restrict__ cbf, const float* __restrict__ cbn,
    _Float16* __restrict__ zq, int* __restrict__ codes,
    float* __restrict__ lossAcc) {
  __shared__ _Float16 sZ[8][256];
  const int lane = threadIdx.x & 31, wave = threadIdx.x >> 5;
  const int row = blockIdx.x * 8 + wave;       // < 32768
  const int b = row >> 10, pix = row & 1023;   // z is NCHW, HW = 1024
  const size_t zbase = ((size_t)b * 256) * 1024 + pix;

  float zz = 0.f;
#pragma unroll
  for (int j = 0; j < 8; ++j) {
    int d = lane + j * 32;
    _Float16 v = z[zbase + (size_t)d * 1024];
    sZ[wave][d] = v;
    zz += (float)v * (float)v;
  }
  __syncthreads();
#pragma unroll
  for (int off = 16; off; off >>= 1) zz += __shfl_xor(zz, off, 32);

  float bestV = 3.4e38f; int bestC = 0;
  for (int ci = 0; ci < 16; ++ci) {
    int c = ci * 32 + lane;
    const _Float16* cr = cbh + (size_t)c * 256;
    float dot = 0.f;
#pragma unroll 8
    for (int d = 0; d < 256; ++d) dot += (float)sZ[wave][d] * (float)cr[d];
    float v = cbn[c] - 2.f * dot;
    if (v < bestV) { bestV = v; bestC = c; }
  }
#pragma unroll
  for (int off = 16; off; off >>= 1) {
    float ov = __shfl_xor(bestV, off, 32);
    int oc = __shfl_xor(bestC, off, 32);
    if (ov < bestV || (ov == bestV && oc < bestC)) { bestV = ov; bestC = oc; }
  }
  int c = __shfl(bestC, 0, 32);

  const float* q = cbf + (size_t)c * 256;
#pragma unroll
  for (int j = 0; j < 8; ++j) {
    int d = lane + j * 32;
    zq[zbase + (size_t)d * 1024] = (_Float16)q[d];
  }
  if (lane == 0) {
    codes[row] = c;
    atomicAdd(lossAcc, zz + bestV);            // ||z - c*||^2
  }
}

__global__ void finalize_loss(const float* __restrict__ lossAcc, float* __restrict__ out) {
  *out = *lossAcc * (1.25f / 8388608.f);       // (0.25+1)*mean, N = 32*256*1024
}

// -------------------- host-side orchestration --------------------------------
static inline size_t alignup(size_t x) { return (x + 255) & ~(size_t)255; }

extern "C" void kernel_launch(void* const* d_in, const int* in_sizes, int n_in,
                              void* d_out, int out_size, void* d_ws, size_t ws_size,
                              hipStream_t stream) {
  (void)in_sizes; (void)n_in; (void)out_size; (void)ws_size;
  const float* x      = (const float*)d_in[0];
  const float* enc_w1 = (const float*)d_in[1];
  const float* enc_b1 = (const float*)d_in[2];
  const float* enc_w2 = (const float*)d_in[3];
  const float* enc_b2 = (const float*)d_in[4];
  const float* enc_w3 = (const float*)d_in[5];
  const float* enc_b3 = (const float*)d_in[6];
  const float* cbf    = (const float*)d_in[7];
  const float* dec_w1 = (const float*)d_in[8];
  const float* dec_b1 = (const float*)d_in[9];
  const float* dec_w2 = (const float*)d_in[10];
  const float* dec_b2 = (const float*)d_in[11];
  const float* dec_w3 = (const float*)d_in[12];
  const float* dec_b3 = (const float*)d_in[13];

  // d_out: x_rec f32 [6291456] | codes i32 [32768] | q_loss f32 [1]
  float* xrec  = (float*)d_out;
  int*   codes = (int*)(xrec + 6291456);
  float* qloss = (float*)(xrec + 6291456 + 32768);

  uint8_t* ws = (uint8_t*)d_ws;
  size_t off = 0;
  auto takeH = [&](size_t n) { _Float16* p = (_Float16*)(ws + off); off = alignup(off + n * 2); return p; };
  auto takeF = [&](size_t n) { float* p = (float*)(ws + off); off = alignup(off + n * 4); return p; };
  _Float16* actA = takeH(67108864);   // h1 / d2
  _Float16* actB = takeH(33554432);   // h2 / d1
  _Float16* zbuf = takeH(8388608);    // z
  _Float16* zq   = takeH(8388608);    // z_q
  _Float16* wp1 = takeH(2 * 8 * 512);        // K=48->2 slabs, NB=8
  _Float16* wp2 = takeH(64 * 16 * 512);      // K=2048, NB=16
  _Float16* wp3 = takeH(128 * 16 * 512);     // K=4096, NB=16
  _Float16* wt1 = takeH((size_t)4 * 32 * 16 * 512);  // K=1024, NB=16
  _Float16* wt2 = takeH((size_t)4 * 32 * 8 * 512);   // K=1024, NB=8
  _Float16* wt3 = takeH((size_t)4 * 16 * 1 * 512);   // K=512,  NB=1
  _Float16* cbh = takeH(512 * 256);
  float* cbn  = takeF(512);
  float* lacc = takeF(1);

  auto cd = [](int a, int b) { return (a + b - 1) / b; };

  // 1) packing
  pack_conv_fr<<<cd(2 * 8 * 512, 256), 256, 0, stream>>>(enc_w1, wp1, 48, 128, 8, 2 * 8 * 512);
  pack_conv_fr<<<cd(64 * 16 * 512, 256), 256, 0, stream>>>(enc_w2, wp2, 2048, 256, 16, 64 * 16 * 512);
  pack_conv_fr<<<cd(128 * 16 * 512, 256), 256, 0, stream>>>(enc_w3, wp3, 4096, 256, 16, 128 * 16 * 512);
  pack_convt_fr<<<cd(4 * 32 * 16 * 512, 256), 256, 0, stream>>>(dec_w1, wt1, 256, 256, 16, 32 * 16 * 512);
  pack_convt_fr<<<cd(4 * 32 * 8 * 512, 256), 256, 0, stream>>>(dec_w2, wt2, 256, 128, 8, 32 * 8 * 512);
  pack_convt_fr<<<cd(4 * 16 * 1 * 512, 256), 256, 0, stream>>>(dec_w3, wt3, 128, 3, 1, 16 * 1 * 512);
  cb_prep<<<cd(512 * 256, 256), 256, 0, stream>>>(cbf, cbh, cbn, lacc);

  // 2) encoder
  conv_gemm<0, 1, float, _Float16><<<dim3(8192, 2, 1), 256, 0, stream>>>(
      x, wp1, enc_b1, actA, 3, 256, 256, 128, 128, 128, 48, 8, 14, 7);
  conv_gemm<0, 1, _Float16, _Float16><<<dim3(2048, 4, 1), 256, 0, stream>>>(
      actA, wp2, enc_b2, actB, 128, 128, 128, 256, 64, 64, 2048, 16, 12, 6);
  conv_gemm<0, 1, _Float16, _Float16><<<dim3(512, 4, 1), 256, 0, stream>>>(
      actB, wp3, enc_b3, zbuf, 256, 64, 64, 256, 32, 32, 4096, 16, 10, 5);

  // 3) vector quantization
  vq_kernel<<<4096, 256, 0, stream>>>(zbuf, cbh, cbf, cbn, zq, codes, lacc);
  finalize_loss<<<1, 1, 0, stream>>>(lacc, qloss);

  // 4) decoder (lgHW/lgWo describe the parity sub-image)
  conv_gemm<1, 1, _Float16, _Float16><<<dim3(512, 4, 4), 256, 0, stream>>>(
      zq, wt1, dec_b1, actB, 256, 32, 32, 256, 64, 64, 1024, 16, 10, 5);
  conv_gemm<1, 1, _Float16, _Float16><<<dim3(2048, 2, 4), 256, 0, stream>>>(
      actB, wt2, dec_b2, actA, 256, 64, 64, 128, 128, 128, 1024, 8, 12, 6);
  conv_gemm<1, 2, _Float16, float><<<dim3(8192, 1, 4), 256, 0, stream>>>(
      actA, wt3, dec_b3, xrec, 128, 128, 128, 3, 256, 256, 512, 1, 14, 7);
}
